// MotionPredictionHead_19207093748301
// MI455X (gfx1250) — matrix operand order repaired and checked by reading, not compiled, not run on hardware
//
#include <hip/hip_runtime.h>
#include <hip/hip_bf16.h>
#include <math.h>

typedef __bf16 bf16;
typedef __attribute__((ext_vector_type(16))) __bf16 v16bf;
typedef __attribute__((ext_vector_type(8)))  float  v8f;

#define B_  8
#define C_  128
#define N_  65536
#define M_  128
#define H_  256
#define NT  128           // points per tile
#define P1  (C_ + 2)      // LDS pitch for X1 (260B = 65 dwords -> bank stride 1)
#define P2  (H_ + 2)      // LDS pitch for X2 (516B = 129 dwords -> bank stride 1)

// ---------------------------------------------------------------------------
// A-fragment: 16x32 A operand packed in contiguous k order (same order as load_b).
// Lane l holds row M=l&15; lanes 0-15 hold K {0..15}, lanes 16-31 hold
// K {16..31}; 16 contiguous bf16 per lane = one 32B load.
// ---------------------------------------------------------------------------
__device__ __forceinline__ v16bf load_a(const bf16* __restrict__ wp, int tile, int lane) {
    return *(const v16bf*)(wp + (tile * 32 + lane) * 16);
}

// B-fragment (KxN = 32x16, bf16): lane l -> column n = l&15; lanes 0-15 hold
// K=kb*32+0..15, lanes 16-31 hold K=kb*32+16..31 -> 16 contiguous bf16 in a
// [n][k] LDS tile.
__device__ __forceinline__ v16bf load_b(const bf16* __restrict__ Xs, int pitch,
                                        int nt, int kb, int lane) {
    int n     = nt * 16 + (lane & 15);
    int kbase = kb * 32 + ((lane & 16) ? 16 : 0);
    const unsigned int* p = (const unsigned int*)(Xs + n * pitch + kbase); // 4B aligned
    union { unsigned int u[8]; v16bf v; } r;
#pragma unroll
    for (int i = 0; i < 8; ++i) r.u[i] = p[i];
    return r.v;
}

// ---------------------------------------------------------------------------
// Prep: fold BN scale into conv weights, cast to bf16 in A-fragment order,
// and produce the fused bias.  idx -> (tile, lane, e) matches load_a().
// ---------------------------------------------------------------------------
__global__ void pack_w(const float* __restrict__ w, const float* __restrict__ bconv,
                       const float* __restrict__ gamma, const float* __restrict__ beta,
                       const float* __restrict__ mean, const float* __restrict__ var,
                       bf16* __restrict__ wp, float* __restrict__ bias_out,
                       int Cin, int KB) {
    int idx = blockIdx.x * blockDim.x + threadIdx.x;
    int total = (H_ / 16) * KB * 512;
    if (idx < total) {
        int e    = idx & 15;
        int lane = (idx >> 4) & 31;
        int tile = idx >> 9;
        int kb   = tile % KB;
        int ht   = tile / KB;
        int row  = lane & 15;
        int kbase = (lane & 16) ? 16 : 0;
        int klocal = kbase + e;
        int h = ht * 16 + row;
        int c = kb * 32 + klocal;
        float inv = gamma[h] * rsqrtf(var[h] + 1e-5f);
        wp[idx] = (bf16)(w[h * Cin + c] * inv);
    }
    if (idx < H_) {
        float inv = gamma[idx] * rsqrtf(var[idx] + 1e-5f);
        bias_out[idx] = bconv[idx] * inv + beta[idx] - mean[idx] * inv;
    }
}

__global__ void init_ws_kernel(float* __restrict__ box_feat, int* __restrict__ counts) {
    int i = blockIdx.x * blockDim.x + threadIdx.x;
    if (i < B_ * M_ * H_) box_feat[i] = 0.0f;  // ReLU output >= 0, 0 is identity for max
    if (i < B_ * M_)      counts[i]  = 0;
}

// ---------------------------------------------------------------------------
// Fused: conv1 -> BN/ReLU -> conv2 -> BN/ReLU -> per-box segment max.
// One workgroup (8 wave32) per (batch, 128-point tile).
// Wave w owns a 64x64 output tile: h-group = w>>1 (4 h-tiles), n-group = w&1
// (4 n-tiles) -> 16 WMMAs per B-fragment-quad per k-step.
// ---------------------------------------------------------------------------
__global__ __launch_bounds__(256) void fused_conv_pool(
    const float* __restrict__ pf, const int* __restrict__ cid,
    const bf16* __restrict__ w1p, const float* __restrict__ bias1,
    const bf16* __restrict__ w2p, const float* __restrict__ bias2,
    float* __restrict__ box_feat, int* __restrict__ counts)
{
    __shared__ bf16 X1[NT * P1];   // [n][c] bf16 activations, tile 1 (~33 KB)
    __shared__ bf16 X2[NT * P2];   // [n][h] bf16 activations, tile 2 (~66 KB)
    __shared__ int  cids[NT];

    const int t  = threadIdx.x;
    const int b  = blockIdx.y;
    const long n0 = (long)blockIdx.x * NT;
    const float* src = pf + (long)b * C_ * N_ + n0;
    const int n  = t & (NT - 1);
    const int cq = t >> 7;  // 0..1

    // speculative prefetch of the next tile (global_prefetch_b8)
    if (blockIdx.x + 1 < gridDim.x)
        __builtin_prefetch(src + (long)(cq * 64) * N_ + n + NT, 0, 1);

    // load + transpose + fp32->bf16: coalesced along n
#pragma unroll 4
    for (int c = cq; c < C_; c += 2)
        X1[n * P1 + c] = (bf16)src[(long)c * N_ + n];
    if (t < NT) cids[t] = cid[(long)b * N_ + n0 + t];
    __syncthreads();

    const int wave = t >> 5;
    const int lane = t & 31;
    const int hg   = wave >> 1;   // 0..3 -> h rows [64*hg, 64*hg+64)
    const int ng   = wave & 1;    // 0..1 -> points [64*ng, 64*ng+64) within tile

    // ---------------- GEMM1: [256x128] x [128x128] ----------------
    v8f acc[4][4];
#pragma unroll
    for (int i = 0; i < 4; ++i)
#pragma unroll
        for (int j = 0; j < 4; ++j) acc[i][j] = v8f{};

#pragma unroll
    for (int kb = 0; kb < C_ / 32; ++kb) {
        v16bf a[4];
#pragma unroll
        for (int i = 0; i < 4; ++i)
            a[i] = load_a(w1p, (hg * 4 + i) * (C_ / 32) + kb, lane);
#pragma unroll
        for (int j = 0; j < 4; ++j) {
            v16bf bb = load_b(X1, P1, ng * 4 + j, kb, lane);
#pragma unroll
            for (int i = 0; i < 4; ++i)
                acc[i][j] = __builtin_amdgcn_wmma_f32_16x16x32_bf16(
                    false, a[i], false, bb, (short)0, acc[i][j], false, false);
        }
    }

    // bias + ReLU + bf16 -> X2 [n][h]
#pragma unroll
    for (int i = 0; i < 4; ++i) {
        int htile = hg * 4 + i;
#pragma unroll
        for (int j = 0; j < 4; ++j) {
            int nn = (ng * 4 + j) * 16 + (lane & 15);
#pragma unroll
            for (int r = 0; r < 8; ++r) {
                int h = htile * 16 + ((lane & 16) ? 8 : 0) + r;
                float val = fmaxf(acc[i][j][r] + bias1[h], 0.0f);
                X2[nn * P2 + h] = (bf16)val;
            }
        }
    }
    __syncthreads();

    // ---------------- GEMM2: [256x256] x [256x128] ----------------
    v8f acc2[4][4];
#pragma unroll
    for (int i = 0; i < 4; ++i)
#pragma unroll
        for (int j = 0; j < 4; ++j) acc2[i][j] = v8f{};

#pragma unroll
    for (int kb = 0; kb < H_ / 32; ++kb) {
        v16bf a[4];
#pragma unroll
        for (int i = 0; i < 4; ++i)
            a[i] = load_a(w2p, (hg * 4 + i) * (H_ / 32) + kb, lane);
#pragma unroll
        for (int j = 0; j < 4; ++j) {
            v16bf bb = load_b(X2, P2, ng * 4 + j, kb, lane);
#pragma unroll
            for (int i = 0; i < 4; ++i)
                acc2[i][j] = __builtin_amdgcn_wmma_f32_16x16x32_bf16(
                    false, a[i], false, bb, (short)0, acc2[i][j], false, false);
        }
    }

    // bias + ReLU + segment max (post-ReLU >= 0 so int-compare of IEEE bits is
    // order-preserving; box_feat pre-initialized to 0.0f)
    float* bf_base = box_feat + (long)b * M_ * H_;
#pragma unroll
    for (int i = 0; i < 4; ++i) {
        int htile = hg * 4 + i;
#pragma unroll
        for (int j = 0; j < 4; ++j) {
            int nn = (ng * 4 + j) * 16 + (lane & 15);
            int cc = cids[nn];
            int* dst = (int*)(bf_base + (long)cc * H_);
#pragma unroll
            for (int r = 0; r < 8; ++r) {
                int h = htile * 16 + ((lane & 16) ? 8 : 0) + r;
                float val = fmaxf(acc2[i][j][r] + bias2[h], 0.0f);
                atomicMax(dst + h, __float_as_int(val));
            }
        }
    }
    if (t < NT) atomicAdd(&counts[b * M_ + cids[t]], 1);
}

// ---------------------------------------------------------------------------
// Per-box MLP: Linear(263->128) ReLU Linear(128->64) ReLU Linear(64->3)
// One 128-thread block per box.
// ---------------------------------------------------------------------------
__global__ __launch_bounds__(128) void box_mlp(
    const float* __restrict__ box_feat, const float* __restrict__ boxes,
    const int* __restrict__ counts,
    const float* __restrict__ fw1, const float* __restrict__ fb1,
    const float* __restrict__ fw2, const float* __restrict__ fb2,
    const float* __restrict__ fw3, const float* __restrict__ fb3,
    float* __restrict__ out)
{
    const int bm = blockIdx.x;           // b*M + m
    const int t  = threadIdx.x;
    __shared__ float comb[H_ + 7];
    __shared__ float h1[128];
    __shared__ float h2[64];

    comb[t]        = box_feat[(long)bm * H_ + t];
    comb[128 + t]  = box_feat[(long)bm * H_ + 128 + t];
    if (t < 7) comb[H_ + t] = boxes[bm * 7 + t];
    __syncthreads();

    {
        float s = fb1[t];
        const float* wr = fw1 + t * (H_ + 7);
#pragma unroll 8
        for (int i = 0; i < H_ + 7; ++i) s += wr[i] * comb[i];
        h1[t] = fmaxf(s, 0.0f);
    }
    __syncthreads();
    if (t < 64) {
        float s = fb2[t];
        const float* wr = fw2 + t * 128;
#pragma unroll 8
        for (int i = 0; i < 128; ++i) s += wr[i] * h1[i];
        h2[t] = fmaxf(s, 0.0f);
    }
    __syncthreads();
    if (t < 3) {
        float s = fb3[t];
        const float* wr = fw3 + t * 64;
#pragma unroll 8
        for (int i = 0; i < 64; ++i) s += wr[i] * h2[i];
        out[bm * 3 + t] = (counts[bm] > 0) ? s : 0.0f;
    }
}

// ---------------------------------------------------------------------------
extern "C" void kernel_launch(void* const* d_in, const int* in_sizes, int n_in,
                              void* d_out, int out_size, void* d_ws, size_t ws_size,
                              hipStream_t stream) {
    const float* pf    = (const float*)d_in[0];
    const float* boxes = (const float*)d_in[1];
    const int*   cid   = (const int*)  d_in[2];
    const float* w1  = (const float*)d_in[3];
    const float* b1  = (const float*)d_in[4];
    const float* g1  = (const float*)d_in[5];
    const float* be1 = (const float*)d_in[6];
    const float* m1  = (const float*)d_in[7];
    const float* v1  = (const float*)d_in[8];
    const float* w2  = (const float*)d_in[9];
    const float* b2  = (const float*)d_in[10];
    const float* g2  = (const float*)d_in[11];
    const float* be2 = (const float*)d_in[12];
    const float* m2  = (const float*)d_in[13];
    const float* v2  = (const float*)d_in[14];
    const float* fw1 = (const float*)d_in[15];
    const float* fb1 = (const float*)d_in[16];
    const float* fw2 = (const float*)d_in[17];
    const float* fb2 = (const float*)d_in[18];
    const float* fw3 = (const float*)d_in[19];
    const float* fb3 = (const float*)d_in[20];
    (void)in_sizes; (void)n_in; (void)out_size; (void)ws_size;

    // workspace layout
    char* ws = (char*)d_ws;
    bf16*  w1p      = (bf16*)(ws + 0);                 //  65536 B
    bf16*  w2p      = (bf16*)(ws + 65536);             // 131072 B
    float* bias1    = (float*)(ws + 196608);           //   1024 B
    float* bias2    = (float*)(ws + 197632);           //   1024 B
    float* box_feat = (float*)(ws + 198656);           // 1048576 B
    int*   counts   = (int*)  (ws + 198656 + 1048576); //   4096 B

    float* out = (float*)d_out;

    init_ws_kernel<<<(B_ * M_ * H_) / 256, 256, 0, stream>>>(box_feat, counts);
    pack_w<<<(16 * 4 * 512) / 256, 256, 0, stream>>>(w1, b1, g1, be1, m1, v1,
                                                     w1p, bias1, C_, C_ / 32);
    pack_w<<<(16 * 8 * 512) / 256, 256, 0, stream>>>(w2, b2, g2, be2, m2, v2,
                                                     w2p, bias2, H_, H_ / 32);

    dim3 grid(N_ / NT, B_);
    fused_conv_pool<<<grid, 256, 0, stream>>>(pf, cid, w1p, bias1, w2p, bias2,
                                              box_feat, counts);

    box_mlp<<<B_ * M_, 128, 0, stream>>>(box_feat, boxes, counts,
                                         fw1, fb1, fw2, fb2, fw3, fb3, out);
}
